// QAMultiheadAttention_50706383897082
// MI455X (gfx1250) — matrix operand order repaired and checked by reading, not compiled, not run on hardware
//
#include <hip/hip_runtime.h>
#include <hip/hip_bf16.h>

typedef __attribute__((ext_vector_type(16))) _Float16 v16h;
typedef __attribute__((ext_vector_type(8)))  float    v8f;

#define EPSF    1e-6f
#define BN_EPSF 1e-5f
#define Bc 4
#define Hc 8
#define Lc 128
#define Dc 64

// workspace layout (float offsets)
#define OFF_PQ   0            // B*H*L*D = 262144
#define OFF_PK   262144
#define OFF_HI   524288
#define OFF_J    786432       // B*H*L*L = 524288
#define OFF_HD   1310720      // 262144
#define OFF_H    1572864      // 4096
#define OFF_HP   1576960      // 4096 (h after BN)
#define OFF_X    1581056      // 4096
#define OFF_ROW  1585152      // 4096 off-diag row sums (atomic, zeroed)
#define OFF_ST   1589248      // 48 stats (atomic, zeroed):
                              // [0..7] h sum, [8..15] h sumsq,
                              // [16..23] J offdiag sum, [24..31] J offdiag sumsq,
                              // [32..39] Hd sum, [40..47] Hd sumsq

// -------- Kernel 1: softmax(Q), softmax(K), V entropy, h stats. wave per row.
__global__ void k_rowprep(const float* __restrict__ Q, const float* __restrict__ K,
                          const float* __restrict__ V, float* __restrict__ ws) {
  const int lane = threadIdx.x & 31;
  const int row  = blockIdx.x * 4 + (threadIdx.x >> 5);   // 0..4095
  const int base = row * Dc;

  // softmax(Q)
  {
    float a0 = Q[base + lane], a1 = Q[base + lane + 32];
    float m = fmaxf(a0, a1);
    #pragma unroll
    for (int s = 16; s > 0; s >>= 1) m = fmaxf(m, __shfl_xor(m, s, 32));
    float e0 = __expf(a0 - m), e1 = __expf(a1 - m);
    float sum = e0 + e1;
    #pragma unroll
    for (int s = 16; s > 0; s >>= 1) sum += __shfl_xor(sum, s, 32);
    float r = 1.0f / sum;
    ws[OFF_PQ + base + lane]      = e0 * r;
    ws[OFF_PQ + base + lane + 32] = e1 * r;
  }
  // softmax(K)
  {
    float a0 = K[base + lane], a1 = K[base + lane + 32];
    float m = fmaxf(a0, a1);
    #pragma unroll
    for (int s = 16; s > 0; s >>= 1) m = fmaxf(m, __shfl_xor(m, s, 32));
    float e0 = __expf(a0 - m), e1 = __expf(a1 - m);
    float sum = e0 + e1;
    #pragma unroll
    for (int s = 16; s > 0; s >>= 1) sum += __shfl_xor(sum, s, 32);
    float r = 1.0f / sum;
    ws[OFF_PK + base + lane]      = e0 * r;
    ws[OFF_PK + base + lane + 32] = e1 * r;
  }
  // V entropy
  {
    float v0 = V[base + lane], v1 = V[base + lane + 32];
    float vs = v0 + v1;
    #pragma unroll
    for (int s = 16; s > 0; s >>= 1) vs += __shfl_xor(vs, s, 32);
    float r = 1.0f / (vs + EPSF);
    float vn0 = fminf(fmaxf(v0 * r, EPSF), 1.0f);
    float vn1 = fminf(fmaxf(v1 * r, EPSF), 1.0f);
    float hi0 = -vn0 * __log2f(vn0);
    float hi1 = -vn1 * __log2f(vn1);
    ws[OFF_HI + base + lane]      = hi0;
    ws[OFF_HI + base + lane + 32] = hi1;
    float hs = hi0 + hi1;
    #pragma unroll
    for (int s = 16; s > 0; s >>= 1) hs += __shfl_xor(hs, s, 32);
    if (lane == 0) {
      ws[OFF_H + row] = hs;
      int head = (row >> 7) & 7;
      atomicAdd(&ws[OFF_ST + head],     hs);
      atomicAdd(&ws[OFF_ST + 8 + head], hs * hs);
    }
  }
}

// -------- Kernel 2: J tiles (JSD) + off-diag per-head sum/sumsq + row sums.
// grid (tj=8, ti=8, bh=32), block 256 = one 16x16 tile.
// pQ/pK tiles staged via ASYNC global->LDS copies (one b128 per lane = 4KB/tile),
// keeping the VALU free for the transcendental-heavy JSD inner loop.
__global__ void k_jsd(float* __restrict__ ws) {
  __shared__ float sQ[16][64];
  __shared__ float sK[16][64];
  __shared__ float jt[16][16];
  __shared__ float red[256];
  const int bh  = blockIdx.z;
  const int ti  = blockIdx.y;
  const int tj  = blockIdx.x;
  const int tid = threadIdx.x;

  {
    const float* srcQ = ws + OFF_PQ + ((bh * Lc) + ti * 16) * Dc;  // 16x64 f32 = 4KB
    const float* srcK = ws + OFF_PK + ((bh * Lc) + tj * 16) * Dc;
    unsigned voff = (unsigned)(tid * 16);                           // byte offset per lane
    unsigned ldsQ = (unsigned)(uintptr_t)(&sQ[0][0]) + voff;
    unsigned ldsK = (unsigned)(uintptr_t)(&sK[0][0]) + voff;
    asm volatile("global_load_async_to_lds_b128 %0, %1, %2"
                 :: "v"(ldsQ), "v"(voff), "s"(srcQ) : "memory");
    asm volatile("global_load_async_to_lds_b128 %0, %1, %2"
                 :: "v"(ldsK), "v"(voff), "s"(srcK) : "memory");
    asm volatile("s_wait_asynccnt 0x0" ::: "memory");
  }
  __syncthreads();

  const int li = tid >> 4, lj = tid & 15;
  float acc = 0.0f;
  #pragma unroll 4
  for (int d = 0; d < 64; ++d) {
    float pq = sQ[li][d], pk = sK[lj][d];
    float m  = 0.5f * (pq + pk);
    float rm = 1.0f / (m + EPSF);
    acc += pq * __log2f(pq * rm + EPSF) + pk * __log2f(pk * rm + EPSF);
  }
  float Jv = 0.5f * acc;
  const int i = ti * 16 + li, j = tj * 16 + lj;
  ws[OFF_J + bh * (Lc * Lc) + i * Lc + j] = Jv;

  float od = (i != j) ? Jv : 0.0f;
  jt[li][lj] = od;

  red[tid] = od;
  __syncthreads();
  for (int s = 128; s > 0; s >>= 1) { if (tid < s) red[tid] += red[tid + s]; __syncthreads(); }
  if (tid == 0) atomicAdd(&ws[OFF_ST + 16 + (bh & 7)], red[0]);
  __syncthreads();
  red[tid] = od * od;
  __syncthreads();
  for (int s = 128; s > 0; s >>= 1) { if (tid < s) red[tid] += red[tid + s]; __syncthreads(); }
  if (tid == 0) atomicAdd(&ws[OFF_ST + 24 + (bh & 7)], red[0]);

  if (lj == 0) {
    float rs = 0.0f;
    #pragma unroll
    for (int c = 0; c < 16; ++c) rs += jt[li][c];
    atomicAdd(&ws[OFF_ROW + bh * Lc + i], rs);
  }
}

// -------- Kernel 3: finalize BN coeffs, h' = a*h+c, gate x via affine row-sum trick.
__global__ void k_gate(float* __restrict__ ws, const float* __restrict__ jg,
                       const float* __restrict__ jb, const float* __restrict__ hg,
                       const float* __restrict__ hb) {
  int idx = blockIdx.x * 256 + threadIdx.x;     // 0..4095
  int bh = idx >> 7, i = idx & 127, head = bh & 7;
  const float* st = ws + OFF_ST;

  float nh = (float)(Bc * Lc);
  float mh = st[head] / nh;
  float vh = st[8 + head] / nh - mh * mh;
  float ah = hg[head] * rsqrtf(vh + BN_EPSF);
  float ch = hb[head] - mh * ah;
  float hp = ah * ws[OFF_H + idx] + ch;
  ws[OFF_HP + idx] = hp;

  float nj = (float)(Bc * Lc * (Lc - 1));
  float mj = st[16 + head] / nj;
  float vj = st[24 + head] / nj - mj * mj;
  float aj = jg[head] * rsqrtf(vj + BN_EPSF);
  float cj = jb[head] - mj * aj;

  float Jd = ws[OFF_J + bh * (Lc * Lc) + i * (Lc + 1)];
  // sum_j J'[i,j] = aj * offdiag_rowsum + 127*cj + J[i,i]
  float rsum = aj * ws[OFF_ROW + idx] + 127.0f * cj + Jd;
  ws[OFF_X + idx] = (rsum + hp < 0.0f) ? 1.0f : 0.0f;
}

// -------- Kernel 4: WMMA matvec y = J x (f16 inputs, f32 accum), He, Hd prepass + stats.
// one block (8 waves) per (b,h); wave w owns rows [16w, 16w+16).
__global__ void k_matvec(float* __restrict__ ws, const float* __restrict__ jg,
                         const float* __restrict__ jb) {
  __shared__ float xs[128];
  __shared__ float hps[128];
  __shared__ float ys[128];
  __shared__ float hes[128];
  __shared__ float Ssh;
  __shared__ float red[256];
  const int bh = blockIdx.x, tid = threadIdx.x, head = bh & 7;
  const float* Jp = ws + OFF_J + bh * (Lc * Lc);

  // prefetch this wave's 16x128 J row-block while we stage x / h'
  {
    const int wave = tid >> 5, lane = tid & 31;
    const float* jrow = Jp + (wave * 16 + (lane & 15)) * Lc + (lane >> 4) * 64;
    __builtin_prefetch(jrow, 0, 3);
  }

  if (tid < 128) {
    xs[tid]  = ws[OFF_X + bh * Lc + tid];
    hps[tid] = ws[OFF_HP + bh * Lc + tid];
  }
  __syncthreads();
  if (tid == 0) {
    float s = 0.0f;
    for (int j = 0; j < 128; ++j) s += xs[j];
    Ssh = s;
  }

  // ---- WMMA: y_tile(16) accumulated over K in 4 chunks of 32 (f16 A/B, f32 C/D)
  const int wave = tid >> 5, lane = tid & 31;
  const int mrow  = wave * 16 + (lane & 15);
  const int akoff = (lane >> 4) * 8;    // A: upper half-wave holds K+8..15 of each 16-group
  const int bkoff = (lane >> 4) * 8;    // B: upper half-wave holds K+8..15 of each 16-group
  const float bsel = ((lane & 15) == 0) ? 1.0f : 0.0f;  // x lives in column N==0
  v8f acc = {};
  for (int kc = 0; kc < 4; ++kc) {
    const int kb = kc * 32;
    v16h a, b;
    #pragma unroll
    for (int p = 0; p < 8; ++p) {
      a[p]     = (_Float16)Jp[mrow * Lc + kb + akoff + p];
      a[8 + p] = (_Float16)Jp[mrow * Lc + kb + 16 + akoff + p];
    }
    #pragma unroll
    for (int q = 0; q < 8; ++q) {
      b[q]     = (_Float16)(bsel * xs[kb + bkoff + q]);
      b[8 + q] = (_Float16)(bsel * xs[kb + 16 + bkoff + q]);
    }
    acc = __builtin_amdgcn_wmma_f32_16x16x32_f16(
        /*neg_a=*/false, a, /*neg_b=*/false, b,
        /*c_mod=*/(short)0, acc, /*reuse_a=*/false, /*reuse_b=*/false);
  }
  // D column 0: lane 0 holds M=0..7 in VGPR 0..7; lane 16 holds M=8..15.
  if ((lane & 15) == 0) {
    const int rbase = wave * 16 + ((lane >> 4) ? 8 : 0);
    #pragma unroll
    for (int p = 0; p < 8; ++p) ys[rbase + p] = acc[p];
  }
  __syncthreads();

  // affine-correct matvec, He
  const float* st = ws + OFF_ST;
  float nj = (float)(Bc * Lc * (Lc - 1));
  float mj = st[16 + head] / nj;
  float vj = st[24 + head] / nj - mj * mj;
  float aj = jg[head] * rsqrtf(vj + BN_EPSF);
  float cj = jb[head] - mj * aj;
  if (tid < 128) {
    float Jd = Jp[tid * (Lc + 1)];
    // sum_j J'[i,j] x_j = aj*(Jx)_i + cj*S + x_i*((1-aj)*J_ii - cj)
    float yp = aj * ys[tid] + cj * Ssh + xs[tid] * ((1.0f - aj) * Jd - cj);
    hes[tid] = xs[tid] * (yp + hps[tid]);
  }
  __syncthreads();

  // Hd_pre = -(He * hi / h'), stream per-head stats
  float lsum = 0.0f, lss = 0.0f;
  for (int e = tid; e < Lc * Dc; e += 256) {
    int i = e >> 6, d = e & 63;
    float hd = -(hes[i] * ws[OFF_HI + (bh * Lc + i) * Dc + d] / hps[i]);
    ws[OFF_HD + (bh * Lc + i) * Dc + d] = hd;
    lsum += hd;
    lss  += hd * hd;
  }
  red[tid] = lsum;
  __syncthreads();
  for (int s = 128; s > 0; s >>= 1) { if (tid < s) red[tid] += red[tid + s]; __syncthreads(); }
  if (tid == 0) atomicAdd(&ws[OFF_ST + 32 + head], red[0]);
  __syncthreads();
  red[tid] = lss;
  __syncthreads();
  for (int s = 128; s > 0; s >>= 1) { if (tid < s) red[tid] += red[tid + s]; __syncthreads(); }
  if (tid == 0) atomicAdd(&ws[OFF_ST + 40 + head], red[0]);
}

// -------- Kernel 5: apply final BN to Hd_pre -> out
__global__ void k_finalbn(const float* __restrict__ ws, const float* __restrict__ g,
                          const float* __restrict__ b, float* __restrict__ out) {
  int idx = blockIdx.x * 256 + threadIdx.x;   // 0..262143
  int head = (idx >> 13) & 7;
  const float* st = ws + OFF_ST;
  float n = (float)(Bc * Lc * Dc);
  float m = st[32 + head] / n;
  float v = st[40 + head] / n - m * m;
  float a = g[head] * rsqrtf(v + BN_EPSF);
  float c = b[head] - m * a;
  out[idx] = a * ws[OFF_HD + idx] + c;
}

extern "C" void kernel_launch(void* const* d_in, const int* in_sizes, int n_in,
                              void* d_out, int out_size, void* d_ws, size_t ws_size,
                              hipStream_t stream) {
  (void)in_sizes; (void)n_in; (void)out_size; (void)ws_size;
  const float* Q  = (const float*)d_in[0];
  const float* K  = (const float*)d_in[1];
  const float* V  = (const float*)d_in[2];
  const float* jg = (const float*)d_in[3];
  const float* jb = (const float*)d_in[4];
  const float* hg = (const float*)d_in[5];
  const float* hb = (const float*)d_in[6];
  const float* bg = (const float*)d_in[7];
  const float* bb = (const float*)d_in[8];
  float* ws  = (float*)d_ws;
  float* out = (float*)d_out;

  // zero atomic accumulators (row sums + stats), graph-capture safe
  hipMemsetAsync(ws + OFF_ROW, 0, (4096 + 48) * sizeof(float), stream);

  k_rowprep<<<dim3(1024), dim3(128), 0, stream>>>(Q, K, V, ws);
  k_jsd    <<<dim3(8, 8, 32), dim3(256), 0, stream>>>(ws);
  k_gate   <<<dim3(16), dim3(256), 0, stream>>>(ws, jg, jb, hg, hb);
  k_matvec <<<dim3(32), dim3(256), 0, stream>>>(ws, jg, jb);
  k_finalbn<<<dim3(1024), dim3(256), 0, stream>>>(ws, bg, bb, out);
}
